// ViT_41704132444384
// MI455X (gfx1250) — matrix operand order repaired and checked by reading, not compiled, not run on hardware
//
#include <hip/hip_runtime.h>
#include <hip/hip_bf16.h>
#include <cmath>

// ---------------------------------------------------------------------------
// Types for CDNA5 WMMA (wave32): v_wmma_f32_16x16x32_bf16
// ---------------------------------------------------------------------------
typedef __attribute__((ext_vector_type(16))) __bf16 v16bf;
typedef __attribute__((ext_vector_type(8)))  __bf16 v8bf;
typedef __attribute__((ext_vector_type(8)))  float  v8f;
typedef unsigned short u16;

#define DEV __device__ __forceinline__

DEV u16 f2bf(float f) {                      // round-to-nearest-even f32 -> bf16
  union { float f; unsigned u; } x; x.f = f;
  unsigned u = x.u;
  return (u16)((u + 0x7FFFu + ((u >> 16) & 1u)) >> 16);
}
DEV float gelu_exact(float v) { return 0.5f * v * (1.0f + erff(v * 0.70710678118654752440f)); }
DEV int imin_(int a, int b) { return a < b ? a : b; }

// ---------------------------------------------------------------------------
// Problem constants
// ---------------------------------------------------------------------------
static constexpr int BATCH = 32;
static constexpr int NVAL  = 197;        // real tokens
static constexpr int NPAD  = 224;        // padded tokens (7*32: works as K of attn@v)
static constexpr int DIMC  = 768;
static constexpr int MLPC  = 3072;
static constexpr int HEADS = 12;
static constexpr int DH    = 64;
static constexpr int DEPTH = 8;
static constexpr int NCLS  = 1000;
static constexpr int QKVC  = 3 * DIMC;   // 2304
static constexpr long long MROWS = (long long)BATCH * NPAD;   // 7168

// ---------------------------------------------------------------------------
// Generic batched WMMA GEMM:  C[m,n] = act( scale*sum_k A[m,k]*W[n,k] + bias[n] + res[m,n] )
// Each wave computes a 32(M) x 64(N) tile: 2 A fragments x 4 B fragments ->
// 8 v_wmma per K-step of 32, so each B fragment feeds two WMMAs (L2-BW relief).
// All M values in this model are multiples of 32. A,B bf16; C f32 or bf16.
// ---------------------------------------------------------------------------
struct GemmArgs {
  const u16* A; const u16* B;
  const float* bias; const float* res;     // res shares C's indexing (same layout)
  float* Cf; u16* Cb;                      // exactly one non-null
  long long lda, ldb, ldc;
  long long a_os, a_is, b_os, b_is, c_os, c_is;  // outer/inner batch strides (elems)
  int inner;                               // inner batch count; gridDim.y = outer*inner
  int M, N, K;
  int nclamp;                              // clamp for B row index (valid N rows in B)
  float scale;
  int act_gelu;
  int rdiv, rstride, rbase;                // row remap: orow = (r/rdiv)*rstride + rbase + r%rdiv
};

__global__ __launch_bounds__(256) void gemm_wmma_bf16(GemmArgs p) {
  const int lane = threadIdx.x & 31;
  const int wid  = threadIdx.x >> 5;
  const int tilesN = (p.N + 63) >> 6;
  const int tilesM = p.M >> 5;                     // M is always a multiple of 32
  const int w = blockIdx.x * 8 + wid;
  if (w >= tilesM * tilesN) return;                // uniform per-wave exit (EXEC stays full)
  const int tn = w % tilesN, tm = w / tilesN;
  const int m0 = tm << 5, n0 = tn << 6;
  const int ob = blockIdx.y / p.inner;
  const int ib = blockIdx.y % p.inner;
  const u16* Ab = p.A + (long long)ob * p.a_os + (long long)ib * p.a_is;
  const u16* Bb = p.B + (long long)ob * p.b_os + (long long)ib * p.b_is;
  const long long cbo = (long long)ob * p.c_os + (long long)ib * p.c_is;

  // A fragment addressing (16-bit A 16x32 layout): lanes 0-15 -> K {0..7,16..23},
  // lanes 16-31 -> K {8..15,24..31}; row = m + (lane&15).
  const int nl   = lane & 15;
  const int ak   = (lane >> 4) << 3;   // 0 or 8
  const int bk   = (lane >> 4) << 3;   // 0 or 8 (B 32x16: lanes 0-15 K {0..7,16..23}, 16-31 K {8..15,24..31})
  const u16* Ap0 = Ab + (long long)(m0 + nl) * p.lda + ak;
  const u16* Ap1 = Ap0 + 16ll * p.lda;
  const long long br0 = (long long)imin_(n0 + nl,      p.nclamp - 1) * p.ldb;
  const long long br1 = (long long)imin_(n0 + nl + 16, p.nclamp - 1) * p.ldb;
  const long long br2 = (long long)imin_(n0 + nl + 32, p.nclamp - 1) * p.ldb;
  const long long br3 = (long long)imin_(n0 + nl + 48, p.nclamp - 1) * p.ldb;

  v8f c00 = {}, c01 = {}, c02 = {}, c03 = {};
  v8f c10 = {}, c11 = {}, c12 = {}, c13 = {};

  for (int k0 = 0; k0 < p.K; k0 += 32) {
    v8bf a0lo = *reinterpret_cast<const v8bf*>(Ap0 + k0);
    v8bf a0hi = *reinterpret_cast<const v8bf*>(Ap0 + k0 + 16);
    v8bf a1lo = *reinterpret_cast<const v8bf*>(Ap1 + k0);
    v8bf a1hi = *reinterpret_cast<const v8bf*>(Ap1 + k0 + 16);
    v16bf a0, a1;
    for (int i = 0; i < 8; ++i) { a0[i] = a0lo[i]; a0[i + 8] = a0hi[i]; }
    for (int i = 0; i < 8; ++i) { a1[i] = a1lo[i]; a1[i + 8] = a1hi[i]; }
    const u16* Bk = Bb + k0 + bk;
#define DO_TILE(ACC0, ACC1, BR) { \
      v8bf b0 = *reinterpret_cast<const v8bf*>(Bk + (BR));      \
      v8bf b1 = *reinterpret_cast<const v8bf*>(Bk + (BR) + 16); \
      v16bf bf_;                                                \
      for (int i = 0; i < 8; ++i) { bf_[i] = b0[i]; bf_[i + 8] = b1[i]; } \
      ACC0 = __builtin_amdgcn_wmma_f32_16x16x32_bf16(false, a0, false, bf_, (short)0, ACC0, false, false); \
      ACC1 = __builtin_amdgcn_wmma_f32_16x16x32_bf16(false, a1, false, bf_, (short)0, ACC1, false, false); }
    DO_TILE(c00, c10, br0)
    DO_TILE(c01, c11, br1)
    DO_TILE(c02, c12, br2)
    DO_TILE(c03, c13, br3)
#undef DO_TILE
  }

  // C/D layout: VGPR e -> (M = mbase + e + (lane>=16 ? 8:0), N = n0t + lane&15)
  const bool hasB = (p.bias != nullptr), hasR = (p.res != nullptr);
  const int rsel = (lane >> 4) << 3;
#define STORE_TILE(ACC, TOFF, MOFF) { \
    int col = n0 + nl + (TOFF); \
    if (col < p.N) { \
      for (int e = 0; e < 8; ++e) { \
        int row = m0 + (MOFF) + rsel + e; \
        long long orow = row; \
        if (p.rdiv > 0) orow = (long long)(row / p.rdiv) * p.rstride + p.rbase + (row % p.rdiv); \
        long long off = cbo + orow * p.ldc + col; \
        float v = ACC[e] * p.scale; \
        if (hasB) v += p.bias[col]; \
        if (hasR) v += p.res[off]; \
        if (p.act_gelu) v = gelu_exact(v); \
        if (p.Cb) p.Cb[off] = f2bf(v); else p.Cf[off] = v; \
      } } }
  STORE_TILE(c00, 0,  0)
  STORE_TILE(c01, 16, 0)
  STORE_TILE(c02, 32, 0)
  STORE_TILE(c03, 48, 0)
  STORE_TILE(c10, 0,  16)
  STORE_TILE(c11, 16, 16)
  STORE_TILE(c12, 32, 16)
  STORE_TILE(c13, 48, 16)
#undef STORE_TILE
}

// ---------------------------------------------------------------------------
// Elementwise / helper kernels
// ---------------------------------------------------------------------------
__global__ void gate_bf16_kernel(const float* w, const float* wm, u16* o, long long n) {
  long long i = (long long)blockIdx.x * blockDim.x + threadIdx.x;
  if (i < n) o[i] = (wm[i] > 0.0f) ? f2bf(w[i]) : (u16)0;   // sigmoid(m)>0.5 <=> m>0
}
__global__ void gate_f32_kernel(const float* b, const float* bm, float* o, long long n) {
  long long i = (long long)blockIdx.x * blockDim.x + threadIdx.x;
  if (i < n) o[i] = (bm[i] > 0.0f) ? b[i] : 0.0f;
}
__global__ void f32_to_bf16_kernel(const float* s, u16* o, long long n) {
  long long i = (long long)blockIdx.x * blockDim.x + threadIdx.x;
  if (i < n) o[i] = f2bf(s[i]);
}

// im2col of 16x16 patches: patches[B*196, 768], d = (pi*16+pj)*3 + c
__global__ void im2col_kernel(const float* x, u16* out) {
  long long idx = (long long)blockIdx.x * blockDim.x + threadIdx.x;
  const long long total = (long long)BATCH * 196 * DIMC;
  if (idx >= total) return;
  int d = (int)(idx % DIMC);
  long long m = idx / DIMC;
  int pch = (int)(m % 196);
  long long b = m / 196;
  int py = pch / 14, px = pch % 14;
  int c = d % 3, pj = (d / 3) % 16, pi = d / 48;
  float v = x[((b * 3 + c) * 224 + (py * 16 + pi)) * 224 + (px * 16 + pj)];
  out[idx] = f2bf(v);
}

// Z[b,0,:] = cls + pos[0]; Z[b,1..196,:] += pos[t]; Z[b,197..223,:] = 0
__global__ void clspos_kernel(const float* cls, const float* pos, float* Z) {
  long long idx = (long long)blockIdx.x * blockDim.x + threadIdx.x;
  const long long total = MROWS * DIMC;
  if (idx >= total) return;
  int d = (int)(idx % DIMC);
  long long r = idx / DIMC;
  int t = (int)(r % NPAD);
  if (t == 0)           Z[idx] = cls[d] + pos[d];
  else if (t < NVAL)    Z[idx] += pos[(long long)t * DIMC + d];
  else                  Z[idx] = 0.0f;
}

// LayerNorm per row (768), bf16 out; padded rows -> 0
__global__ __launch_bounds__(256) void ln_kernel(const float* Z, const float* g,
                                                 const float* b, u16* Y) {
  long long row = blockIdx.x;
  int t = (int)(row % NPAD);
  const float* zr = Z + row * DIMC;
  u16* yr = Y + row * DIMC;
  int tid = threadIdx.x;
  if (t >= NVAL) { for (int i = tid; i < DIMC; i += 256) yr[i] = 0; return; }
  __shared__ float sbuf[256];
  float v0 = zr[tid], v1 = zr[tid + 256], v2 = zr[tid + 512];
  sbuf[tid] = v0 + v1 + v2; __syncthreads();
  for (int o = 128; o > 0; o >>= 1) { if (tid < o) sbuf[tid] += sbuf[tid + o]; __syncthreads(); }
  float mu = sbuf[0] * (1.0f / DIMC); __syncthreads();
  float d0 = v0 - mu, d1 = v1 - mu, d2 = v2 - mu;
  sbuf[tid] = d0 * d0 + d1 * d1 + d2 * d2; __syncthreads();
  for (int o = 128; o > 0; o >>= 1) { if (tid < o) sbuf[tid] += sbuf[tid + o]; __syncthreads(); }
  float rstd = rsqrtf(sbuf[0] * (1.0f / DIMC) + 1e-5f);
  yr[tid]       = f2bf(d0 * rstd * g[tid]       + b[tid]);
  yr[tid + 256] = f2bf(d1 * rstd * g[tid + 256] + b[tid + 256]);
  yr[tid + 512] = f2bf(d2 * rstd * g[tid + 512] + b[tid + 512]);
}

// Row softmax over 224 cols, valid j<197, bf16 out (zeros for padded cols). One wave/row.
__global__ __launch_bounds__(256) void softmax_kernel(const float* S, u16* O, long long rows) {
  int lane = threadIdx.x & 31, wid = threadIdx.x >> 5;
  long long row = (long long)blockIdx.x * 8 + wid;
  if (row >= rows) return;
  const float* sr = S + row * NPAD;
  u16* orp = O + row * NPAD;
  float v[7]; float m = -1e30f;
  for (int i = 0; i < 7; ++i) {
    int j = lane + i * 32;
    float x = (j < NVAL) ? sr[j] : -1e30f;
    v[i] = x; m = fmaxf(m, x);
  }
  for (int o = 16; o > 0; o >>= 1) m = fmaxf(m, __shfl_xor(m, o, 32));
  float s = 0.0f;
  for (int i = 0; i < 7; ++i) {
    int j = lane + i * 32;
    float e = (j < NVAL) ? __expf(v[i] - m) : 0.0f;
    v[i] = e; s += e;
  }
  for (int o = 16; o > 0; o >>= 1) s += __shfl_xor(s, o, 32);
  float inv = 1.0f / s;
  for (int i = 0; i < 7; ++i) orp[lane + i * 32] = f2bf(v[i] * inv);
}

// vT[b,h,d,j] = QKV[b,j, 1536 + h*64 + d]   (so attn@v has contiguous-K B fragments)
__global__ void vtrans_kernel(const u16* QKV, u16* vT) {
  long long idx = (long long)blockIdx.x * blockDim.x + threadIdx.x;
  const long long total = (long long)BATCH * HEADS * DH * NPAD;
  if (idx >= total) return;
  int j = (int)(idx % NPAD);
  long long r = idx / NPAD;
  int d = (int)(r % DH);
  long long bh = r / DH;
  int h = (int)(bh % HEADS);
  long long b = bh / HEADS;
  vT[idx] = QKV[(b * NPAD + j) * QKVC + 2 * DIMC + h * DH + d];
}

__global__ void cls_extract_kernel(const float* Z, u16* cbf) {
  int idx = blockIdx.x * blockDim.x + threadIdx.x;
  if (idx >= BATCH * DIMC) return;
  int d = idx % DIMC, b = idx / DIMC;
  cbf[idx] = f2bf(Z[(long long)b * NPAD * DIMC + d]);
}

// ---------------------------------------------------------------------------
// Host side
// ---------------------------------------------------------------------------
static void launch_gemm(hipStream_t st,
                        const u16* A, long long lda, long long a_os, long long a_is,
                        const u16* B, long long ldb, long long b_os, long long b_is,
                        float* Cf, u16* Cb, long long ldc, long long c_os, long long c_is,
                        int outer, int inner, int M, int N, int K, int nclamp,
                        const float* bias, const float* res, float scale, int gelu,
                        int rdiv, int rstride, int rbase) {
  GemmArgs p;
  p.A = A; p.B = B; p.bias = bias; p.res = res; p.Cf = Cf; p.Cb = Cb;
  p.lda = lda; p.ldb = ldb; p.ldc = ldc;
  p.a_os = a_os; p.a_is = a_is; p.b_os = b_os; p.b_is = b_is; p.c_os = c_os; p.c_is = c_is;
  p.inner = inner; p.M = M; p.N = N; p.K = K; p.nclamp = nclamp;
  p.scale = scale; p.act_gelu = gelu; p.rdiv = rdiv; p.rstride = rstride; p.rbase = rbase;
  int tiles = (M / 32) * ((N + 63) / 64);
  dim3 grid((tiles + 7) / 8, outer * inner);
  gemm_wmma_bf16<<<grid, dim3(256), 0, st>>>(p);
}

extern "C" void kernel_launch(void* const* d_in, const int* in_sizes, int n_in,
                              void* d_out, int out_size, void* d_ws, size_t ws_size,
                              hipStream_t stream) {
  (void)in_sizes; (void)n_in; (void)out_size; (void)ws_size;
  // ---- inputs (all fp32) in setup_inputs() order ----
  const float* x        = (const float*)d_in[0];
  const float* patch_w  = (const float*)d_in[1];
  const float* patch_b  = (const float*)d_in[2];
  const float* cls_tok  = (const float*)d_in[3];
  const float* pos_emb  = (const float*)d_in[4];
  const float* ln1_g    = (const float*)d_in[5];
  const float* ln1_b    = (const float*)d_in[6];
  const float* qkv_w    = (const float*)d_in[7];
  const float* qkv_wm   = (const float*)d_in[8];
  const float* out_w    = (const float*)d_in[9];
  const float* out_wm   = (const float*)d_in[10];
  const float* out_b    = (const float*)d_in[11];
  const float* out_bm   = (const float*)d_in[12];
  const float* ln2_g    = (const float*)d_in[13];
  const float* ln2_b    = (const float*)d_in[14];
  const float* ff1_w    = (const float*)d_in[15];
  const float* ff1_wm   = (const float*)d_in[16];
  const float* ff1_b    = (const float*)d_in[17];
  const float* ff1_bm   = (const float*)d_in[18];
  const float* ff2_w    = (const float*)d_in[19];
  const float* ff2_wm   = (const float*)d_in[20];
  const float* ff2_b    = (const float*)d_in[21];
  const float* ff2_bm   = (const float*)d_in[22];
  const float* c1_w     = (const float*)d_in[23];
  const float* c1_wm    = (const float*)d_in[24];
  const float* c1_b     = (const float*)d_in[25];
  const float* c1_bm    = (const float*)d_in[26];
  const float* c2_w     = (const float*)d_in[27];
  const float* c2_wm    = (const float*)d_in[28];
  const float* c2_b     = (const float*)d_in[29];
  const float* c2_bm    = (const float*)d_in[30];

  // ---- workspace bump allocator (256B aligned segments; ~390 MB total) ----
  char* base = (char*)d_ws; size_t off = 0;
  auto alloc = [&](size_t bytes) -> void* {
    void* p = base + off; off = (off + bytes + 255) & ~(size_t)255; return p;
  };
  const long long nQKVW = (long long)DEPTH * QKVC * DIMC;
  const long long nOUTW = (long long)DEPTH * DIMC * DIMC;
  const long long nFF1W = (long long)DEPTH * MLPC * DIMC;
  const long long nFF2W = (long long)DEPTH * DIMC * MLPC;
  u16*   qkvg  = (u16*)alloc(nQKVW * 2);
  u16*   outg  = (u16*)alloc(nOUTW * 2);
  u16*   ff1g  = (u16*)alloc(nFF1W * 2);
  u16*   ff2g  = (u16*)alloc(nFF2W * 2);
  u16*   c1g   = (u16*)alloc((long long)MLPC * DIMC * 2);
  u16*   c2g   = (u16*)alloc((long long)NCLS * MLPC * 2);
  u16*   pwbf  = (u16*)alloc((long long)DIMC * DIMC * 2);
  float* outbF = (float*)alloc((long long)DEPTH * DIMC * 4);
  float* ff1bF = (float*)alloc((long long)DEPTH * MLPC * 4);
  float* ff2bF = (float*)alloc((long long)DEPTH * DIMC * 4);
  float* c1bF  = (float*)alloc((long long)MLPC * 4);
  float* c2bF  = (float*)alloc((long long)NCLS * 4);
  float* Z     = (float*)alloc(MROWS * DIMC * 4);                              // residual stream
  u16*   Ybf   = (u16*)alloc(MROWS * DIMC * 2);                                // LN output
  u16*   QKVb  = (u16*)alloc(MROWS * QKVC * 2);
  float* Sf    = (float*)alloc((long long)BATCH * HEADS * NPAD * NPAD * 4);    // scores
  u16*   Attnb = (u16*)alloc((long long)BATCH * HEADS * NPAD * NPAD * 2);
  u16*   vTb   = (u16*)alloc((long long)BATCH * HEADS * DH * NPAD * 2);
  u16*   AOb   = (u16*)alloc(MROWS * DIMC * 2);                                // attention output
  u16*   Hb    = (u16*)alloc(MROWS * MLPC * 2);                                // MLP hidden
  u16*   col   = (u16*)alloc((long long)BATCH * 196 * DIMC * 2);               // im2col
  u16*   clsb  = (u16*)alloc((long long)BATCH * DIMC * 2);
  u16*   H1b   = (u16*)alloc((long long)BATCH * MLPC * 2);

  auto blk = [](long long n) { return (unsigned)((n + 255) / 256); };

  // ---- 1. gate weights (sigmoid(m)>0.5 <=> m>0) into bf16, biases into f32 ----
  gate_bf16_kernel<<<blk(nQKVW), 256, 0, stream>>>(qkv_w, qkv_wm, qkvg, nQKVW);
  gate_bf16_kernel<<<blk(nOUTW), 256, 0, stream>>>(out_w, out_wm, outg, nOUTW);
  gate_bf16_kernel<<<blk(nFF1W), 256, 0, stream>>>(ff1_w, ff1_wm, ff1g, nFF1W);
  gate_bf16_kernel<<<blk(nFF2W), 256, 0, stream>>>(ff2_w, ff2_wm, ff2g, nFF2W);
  gate_bf16_kernel<<<blk((long long)MLPC * DIMC), 256, 0, stream>>>(c1_w, c1_wm, c1g, (long long)MLPC * DIMC);
  gate_bf16_kernel<<<blk((long long)NCLS * MLPC), 256, 0, stream>>>(c2_w, c2_wm, c2g, (long long)NCLS * MLPC);
  gate_f32_kernel<<<blk(DEPTH * DIMC), 256, 0, stream>>>(out_b, out_bm, outbF, DEPTH * DIMC);
  gate_f32_kernel<<<blk(DEPTH * MLPC), 256, 0, stream>>>(ff1_b, ff1_bm, ff1bF, DEPTH * MLPC);
  gate_f32_kernel<<<blk(DEPTH * DIMC), 256, 0, stream>>>(ff2_b, ff2_bm, ff2bF, DEPTH * DIMC);
  gate_f32_kernel<<<blk(MLPC), 256, 0, stream>>>(c1_b, c1_bm, c1bF, MLPC);
  gate_f32_kernel<<<blk(NCLS), 256, 0, stream>>>(c2_b, c2_bm, c2bF, NCLS);
  f32_to_bf16_kernel<<<blk((long long)DIMC * DIMC), 256, 0, stream>>>(patch_w, pwbf, (long long)DIMC * DIMC);

  // ---- 2. patch embedding: im2col -> WMMA GEMM scattered into Z rows 1..196 ----
  im2col_kernel<<<blk((long long)BATCH * 196 * DIMC), 256, 0, stream>>>(x, col);
  // remap rstride/rbase are in rows (multiplied by ldc in-kernel): orow = (r/196)*224 + 1 + r%196
  launch_gemm(stream, col, DIMC, 0, 0, pwbf, DIMC, 0, 0,
              Z, nullptr, DIMC, 0, 0, 1, 1,
              BATCH * 196, DIMC, DIMC, DIMC, patch_b, nullptr, 1.0f, 0,
              196, NPAD, 1);
  clspos_kernel<<<blk(MROWS * DIMC), 256, 0, stream>>>(cls_tok, pos_emb, Z);

  const float att_scale = 1.0f / sqrtf((float)DIMC);
  const long long rowsBH = (long long)BATCH * HEADS * NPAD;   // 86016 softmax rows

  // ---- 3. transformer layers ----
  for (int i = 0; i < DEPTH; ++i) {
    ln_kernel<<<(unsigned)MROWS, 256, 0, stream>>>(Z, ln1_g + (long long)i * DIMC,
                                                   ln1_b + (long long)i * DIMC, Ybf);
    // qkv = y @ Wqkv^T  -> bf16 [7168, 2304]
    launch_gemm(stream, Ybf, DIMC, 0, 0, qkvg + (long long)i * QKVC * DIMC, DIMC, 0, 0,
                nullptr, QKVb, QKVC, 0, 0, 1, 1,
                (int)MROWS, QKVC, DIMC, QKVC, nullptr, nullptr, 1.0f, 0, 0, 0, 0);
    // scores = scale * q k^T, batched over (b,h)
    launch_gemm(stream,
                QKVb, QKVC, (long long)NPAD * QKVC, DH,
                QKVb + DIMC, QKVC, (long long)NPAD * QKVC, DH,
                Sf, nullptr, NPAD, (long long)HEADS * NPAD * NPAD, (long long)NPAD * NPAD,
                BATCH, HEADS, NPAD, NPAD, DH, NPAD,
                nullptr, nullptr, att_scale, 0, 0, 0, 0);
    softmax_kernel<<<(unsigned)((rowsBH + 7) / 8), 256, 0, stream>>>(Sf, Attnb, rowsBH);
    vtrans_kernel<<<blk((long long)BATCH * HEADS * DH * NPAD), 256, 0, stream>>>(QKVb, vTb);
    // o = attn @ v  -> AttnOut[b, i, h*64 + d]
    launch_gemm(stream,
                Attnb, NPAD, (long long)HEADS * NPAD * NPAD, (long long)NPAD * NPAD,
                vTb, NPAD, (long long)HEADS * DH * NPAD, (long long)DH * NPAD,
                nullptr, AOb, DIMC, (long long)NPAD * DIMC, DH,
                BATCH, HEADS, NPAD, DH, NPAD, DH,
                nullptr, nullptr, 1.0f, 0, 0, 0, 0);
    // z = z + o @ Wout^T + b
    launch_gemm(stream, AOb, DIMC, 0, 0, outg + (long long)i * DIMC * DIMC, DIMC, 0, 0,
                Z, nullptr, DIMC, 0, 0, 1, 1,
                (int)MROWS, DIMC, DIMC, DIMC, outbF + (long long)i * DIMC, Z, 1.0f, 0, 0, 0, 0);
    ln_kernel<<<(unsigned)MROWS, 256, 0, stream>>>(Z, ln2_g + (long long)i * DIMC,
                                                   ln2_b + (long long)i * DIMC, Ybf);
    // h = gelu(y @ Wff1^T + b)
    launch_gemm(stream, Ybf, DIMC, 0, 0, ff1g + (long long)i * MLPC * DIMC, DIMC, 0, 0,
                nullptr, Hb, MLPC, 0, 0, 1, 1,
                (int)MROWS, MLPC, DIMC, MLPC, ff1bF + (long long)i * MLPC, nullptr, 1.0f, 1, 0, 0, 0);
    // z = z + h @ Wff2^T + b
    launch_gemm(stream, Hb, MLPC, 0, 0, ff2g + (long long)i * DIMC * MLPC, MLPC, 0, 0,
                Z, nullptr, DIMC, 0, 0, 1, 1,
                (int)MROWS, DIMC, MLPC, DIMC, ff2bF + (long long)i * DIMC, Z, 1.0f, 0, 0, 0, 0);
  }

  // ---- 4. classifier head ----
  cls_extract_kernel<<<blk(BATCH * DIMC), 256, 0, stream>>>(Z, clsb);
  launch_gemm(stream, clsb, DIMC, 0, 0, c1g, DIMC, 0, 0,
              nullptr, H1b, MLPC, 0, 0, 1, 1,
              BATCH, MLPC, DIMC, MLPC, c1bF, nullptr, 1.0f, 1, 0, 0, 0);
  launch_gemm(stream, H1b, MLPC, 0, 0, c2g, MLPC, 0, 0,
              (float*)d_out, nullptr, NCLS, 0, 0, 1, 1,
              BATCH, NCLS, MLPC, NCLS, c2bF, nullptr, 1.0f, 0, 0, 0, 0);
}
